// PhotometricLoss_57329223467369
// MI455X (gfx1250) — matrix operand order repaired and checked by reading, not compiled, not run on hardware
//
#include <hip/hip_runtime.h>

typedef __attribute__((ext_vector_type(16))) _Float16 v16h;
typedef __attribute__((ext_vector_type(8)))  float    v8f;

#define BATCH   4
#define IMH     1080
#define IMW     1920
#define NCH     3
#define TILES_H 68          // ceil(1080/16)
#define TILES_W 120         // 1920/16
#define NUM_TILES (BATCH*NCH*TILES_H*TILES_W)
#define SSIM_C1 1.0e-4f
#define SSIM_C2 9.0e-4f

#if __has_builtin(__builtin_amdgcn_global_load_async_to_lds_b32) && \
    __has_builtin(__builtin_amdgcn_s_wait_asynccnt)
#define USE_ASYNC_LDS 1
#else
#define USE_ASYNC_LDS 0
#endif

// Gaussian tap w[j], j in [0,10], sigma=1.5, normalized; 0 outside band.
__device__ __forceinline__ float gw(int j, float invs) {
  float d = (float)(j - 5);
  float v = __expf(-d * d * (1.0f / 4.5f)) * invs;
  return (j >= 0 && j <= 10) ? v : 0.0f;
}

__global__ __launch_bounds__(32)
void ssim_l1_kernel(const float* __restrict__ xin, const float* __restrict__ yin,
                    float* __restrict__ pl1, float* __restrict__ pss)
{
  // One wave per block; each wave owns one 16x16 output tile at a time.
  __shared__ float     sx[32 * 32];         // raw x region (rows>=26 stay zero)
  __shared__ float     sy[32 * 32];         // raw y region
  __shared__ _Float16  hT[5][16 * 32];      // H intermediates, [q][n*32 + k]

  const int lane = threadIdx.x;             // 0..31
  const int half = (lane & 16) ? 1 : 0;
  const int nlo  = lane & 15;

  // ---- normalized Gaussian ----
  float s = 0.f;
#pragma unroll
  for (int i = 0; i < 11; ++i) { float d = (float)(i - 5); s += __expf(-d * d * (1.0f / 4.5f)); }
  const float invs = 1.0f / s;

  // ---- B_h: 32x16 banded filter, B[k][n] = w[k-n].
  // B f16 layout: col N=lane&15; element t: K = 8*half + t (t<8), 16 + 8*half + (t-8) (t>=8).
  v16h bh;
#pragma unroll
  for (int t = 0; t < 16; ++t) {
    int k = (t < 8) ? (8 * half + t) : (16 + 8 * half + (t - 8));
    bh[t] = (_Float16)gw(k - nlo, invs);
  }

  // ---- A_v: 16x32 banded filter, A[m][k] = w[k-m].
  // A f16 in contiguous k order used here (as bv): lane M=lane&15; VGPR v: K = 2v + 16*half.
  v16h av;
#pragma unroll
  for (int vv = 0; vv < 8; ++vv) {
    int k0 = 2 * vv + 16 * half;
    av[2 * vv]     = (_Float16)gw(k0     - nlo, invs);
    av[2 * vv + 1] = (_Float16)gw(k0 + 1 - nlo, invs);
  }

  // Rows 26..31 of the staged region are never written by staging; zero once.
#pragma unroll
  for (int r = 26; r < 32; ++r) {
    sx[r * 32 + lane] = 0.f;
    sy[r * 32 + lane] = 0.f;
  }
  __syncthreads();

  float l1acc = 0.f, ssacc = 0.f;

  for (int tile = blockIdx.x; tile < NUM_TILES; tile += gridDim.x) {
    int t = tile;
    const int txi = t % TILES_W; t /= TILES_W;
    const int tyi = t % TILES_H; t /= TILES_H;
    const int ch  = t % NCH;     t /= NCH;
    const int b   = t;
    const int x0 = txi * 16 - 5;
    const int y0 = tyi * 16 - 5;
    const int bbase = b * (IMH * IMW * NCH) + ch;

    // Interior tiles: whole 26x32 region (incl. weight-zeroed cols 26..31)
    // is strictly inside the image -> unguarded async staging is safe.
    const bool interior = (txi >= 1) && (txi <= 118) && (tyi >= 1) && (tyi <= 66);

#if USE_ASYNC_LDS
    if (interior) {
      // ---- async global -> LDS staging (ASYNCcnt-tracked, no VGPR round trip)
#pragma unroll 2
      for (int r = 0; r < 26; ++r) {
        int gi = bbase + ((y0 + r) * IMW + (x0 + lane)) * NCH;
        __builtin_amdgcn_global_load_async_to_lds_b32(
            (__attribute__((address_space(1))) int*)(uintptr_t)(xin + gi),
            (__attribute__((address_space(3))) int*)(uintptr_t)(&sx[r * 32 + lane]),
            0, 0);
        __builtin_amdgcn_global_load_async_to_lds_b32(
            (__attribute__((address_space(1))) int*)(uintptr_t)(yin + gi),
            (__attribute__((address_space(3))) int*)(uintptr_t)(&sy[r * 32 + lane]),
            0, 0);
      }
      __builtin_amdgcn_s_wait_asynccnt(0);
    } else
#endif
    {
      // ---- guarded sync staging (boundary tiles; SAME zero padding) ----
#pragma unroll 2
      for (int r = 0; r < 26; ++r) {
        int gy = y0 + r;
        int gx = x0 + lane;
        bool ok = ((unsigned)gy < (unsigned)IMH) && ((unsigned)gx < (unsigned)IMW);
        float xv = 0.f, yv = 0.f;
        if (ok) {
          int gi = bbase + (gy * IMW + gx) * NCH;
          xv = xin[gi];
          yv = yin[gi];
        }
        sx[r * 32 + lane] = xv;
        sy[r * 32 + lane] = yv;
      }
    }

    // ---- prefetch next tile's region toward L2/WGP$ while we compute ----
    {
      int tn = tile + gridDim.x;
      if (tn < NUM_TILES) {                  // wave-uniform
        int u = tn;
        int txn = u % TILES_W; u /= TILES_W;
        int tyn = u % TILES_H; u /= TILES_H;
        int chn = u % NCH;     u /= NCH;
        int bn  = u;
        int gy = tyn * 16 - 5 + lane;
        if (lane < 26 && (unsigned)gy < (unsigned)IMH) {
          int gx = txn * 16 - 5; if (gx < 0) gx = 0;
          int gi = bn * (IMH * IMW * NCH) + chn + (gy * IMW + gx) * NCH;
          __builtin_prefetch(&xin[gi]);
          __builtin_prefetch(&yin[gi]);
          __builtin_prefetch(&xin[gi + 48]);   // +16 pixels
          __builtin_prefetch(&yin[gi + 48]);
        }
      }
    }
    __syncthreads();

    // ---- L1 over the central 16x16 (padding zeros contribute 0) ----
#pragma unroll
    for (int i = 0; i < 8; ++i) {
      int e  = lane + 32 * i;            // 0..255
      int rr = 5 + (e >> 4);
      int cc = 5 + (e & 15);
      l1acc += fabsf(sx[rr * 32 + cc] - sy[rr * 32 + cc]);
    }

    // ---- horizontal pass: two row-blocks, 5 quantities each ----
#pragma unroll
    for (int rb = 0; rb < 2; ++rb) {
      const int r  = rb * 16 + nlo;      // region row for this lane's A rows
      const float* rwx = &sx[r * 32];
      const float* rwy = &sy[r * 32];
      const int ko = 8 * half;
      float xs[16], ys[16];
#pragma unroll
      for (int tt = 0; tt < 8; ++tt) {
        xs[tt]     = rwx[ko + tt];       ys[tt]     = rwy[ko + tt];
        xs[8 + tt] = rwx[16 + ko + tt];  ys[8 + tt] = rwy[16 + ko + tt];
      }
      v16h ax, ay, axx, ayy, axy;
#pragma unroll
      for (int tt = 0; tt < 16; ++tt) {
        float xv = xs[tt], yv = ys[tt];
        ax[tt]  = (_Float16)xv;
        ay[tt]  = (_Float16)yv;
        axx[tt] = (_Float16)(xv * xv);
        ayy[tt] = (_Float16)(yv * yv);
        axy[tt] = (_Float16)(xv * yv);
      }
      const int kb = rb * 16 + 8 * half; // H-row base this lane's C rows map to
      v8f cz = {};
      auto hstep = [&](v16h a, int q) {
        v8f c = __builtin_amdgcn_wmma_f32_16x16x32_f16(
            false, a, false, bh, (short)0, cz, false, false);
        _Float16* hp = &hT[q][nlo * 32 + kb];
#pragma unroll
        for (int tt = 0; tt < 8; ++tt) hp[tt] = (_Float16)c[tt];
      };
      hstep(ax, 0); hstep(ay, 1); hstep(axx, 2); hstep(ayy, 3); hstep(axy, 4);
    }
    __syncthreads();

    // ---- vertical pass: banded weights x H ----
    v8f S[5];
    const int ko2 = 16 * half;
#pragma unroll
    for (int q = 0; q < 5; ++q) {
      const _Float16* hp = &hT[q][nlo * 32 + ko2];
      v16h bv;
#pragma unroll
      for (int tt = 0; tt < 16; ++tt) bv[tt] = hp[tt];
      v8f cz = {};
      S[q] = __builtin_amdgcn_wmma_f32_16x16x32_f16(
          false, av, false, bv, (short)0, cz, false, false);
    }

    // ---- SSIM map (mask rows past image bottom) ----
    const int gy0 = tyi * 16 + 8 * half;
#pragma unroll
    for (int vv = 0; vv < 8; ++vv) {
      float mu1 = S[0][vv], mu2 = S[1][vv];
      float exx = S[2][vv], eyy = S[3][vv], exy = S[4][vv];
      float mu1sq = mu1 * mu1, mu2sq = mu2 * mu2, mu12 = mu1 * mu2;
      float s1 = exx - mu1sq, s2 = eyy - mu2sq, s12 = exy - mu12;
      float num = (2.f * mu12 + SSIM_C1) * (2.f * s12 + SSIM_C2);
      float den = (mu1sq + mu2sq + SSIM_C1) * (s1 + s2 + SSIM_C2);
      float v = num / den;
      ssacc += ((gy0 + vv) < IMH) ? v : 0.f;
    }
    __syncthreads();
  }

  // ---- wave reduction, one partial pair per block ----
#pragma unroll
  for (int off = 16; off > 0; off >>= 1) {
    l1acc += __shfl_xor(l1acc, off, 32);
    ssacc += __shfl_xor(ssacc, off, 32);
  }
  if (lane == 0) {
    pl1[blockIdx.x] = l1acc;
    pss[blockIdx.x] = ssacc;
  }
}

__global__ __launch_bounds__(256)
void finalize_kernel(const float* __restrict__ pl1, const float* __restrict__ pss,
                     float* __restrict__ out, int nparts)
{
  __shared__ float a[256];
  __shared__ float c[256];
  float s1 = 0.f, s2 = 0.f;
  for (int i = threadIdx.x; i < nparts; i += 256) { s1 += pl1[i]; s2 += pss[i]; }
  a[threadIdx.x] = s1;
  c[threadIdx.x] = s2;
  __syncthreads();
  for (int off = 128; off > 0; off >>= 1) {
    if ((int)threadIdx.x < off) {
      a[threadIdx.x] += a[threadIdx.x + off];
      c[threadIdx.x] += c[threadIdx.x + off];
    }
    __syncthreads();
  }
  if (threadIdx.x == 0) {
    const float N = (float)(BATCH * IMH * IMW * NCH);   // 24,883,200
    float l1    = a[0] / N;
    float ssimv = c[0] / N;
    float dssim = (1.0f - ssimv) * 0.5f;
    out[0] = 0.8f * l1 + 0.2f * dssim;
    out[1] = l1;
    out[2] = dssim;
  }
}

extern "C" void kernel_launch(void* const* d_in, const int* in_sizes, int n_in,
                              void* d_out, int out_size, void* d_ws, size_t ws_size,
                              hipStream_t stream) {
  (void)in_sizes; (void)n_in; (void)out_size;
  const float* x = (const float*)d_in[0];   // rendered
  const float* y = (const float*)d_in[1];   // target
  float* out = (float*)d_out;

  // partials in workspace; clamp block count to workspace capacity
  int nblocks = 8192;
  size_t need = (size_t)nblocks * 2 * sizeof(float);
  if (ws_size < need) {
    nblocks = (int)(ws_size / (2 * sizeof(float)));
    if (nblocks < 1) nblocks = 1;
  }
  float* pl1 = (float*)d_ws;
  float* pss = pl1 + nblocks;

  ssim_l1_kernel<<<nblocks, 32, 0, stream>>>(x, y, pl1, pss);
  finalize_kernel<<<1, 256, 0, stream>>>(pl1, pss, out, nblocks);
}
